// SplineBlock_43843026158076
// MI455X (gfx1250) — compile-verified, matrix operand order checked by reading, not run on hardware
//
#include <hip/hip_runtime.h>
#include <hip/hip_bf16.h>
#include <math.h>

#define N_NODES 100000
#define E_EDGES 3200000
#define D_IN    16
#define D_OUT   32
#define NUM_K   8
#define BN_EPS  1e-5f

typedef __attribute__((ext_vector_type(16))) _Float16 v16h;
typedef __attribute__((ext_vector_type(2)))  _Float16 h2;
typedef __attribute__((ext_vector_type(8)))  float    v8f;
typedef __attribute__((ext_vector_type(4)))  int      v4i;

// ---------------------------------------------------------------------------
// Kernel 0: zero the accumulators (agg[N*32], cnt[N], gsum[32], gsq[32])
// ---------------------------------------------------------------------------
__global__ void k_zero(float* __restrict__ ws, int n)
{
    int stride = gridDim.x * blockDim.x;
    for (int i = blockIdx.x * blockDim.x + threadIdx.x; i < n; i += stride)
        ws[i] = 0.0f;
}

// ---------------------------------------------------------------------------
// Kernel 1: pack Wf[k*16+i][n] = W[k][i][n] (f32 [8,16,32]) into the exact
// per-lane f16 B-fragment layout for v_wmma_f32_16x16x32_f16.
// Layout: WfB[(t*4+c)*32 + lane][h], t = N-tile (0,1), c = K-chunk (0..3),
// h = half index 0..15 inside the lane's v16h.
//   kb = (lane<16) ? 0 : 8
//   localK = (h<8) ? kb+h : kb+16+(h-8);  K = 32c + localK;  n = 16t + (lane&15)
// ---------------------------------------------------------------------------
__global__ void k_prep(const float* __restrict__ W, _Float16* __restrict__ WfB)
{
    int idx = blockIdx.x * blockDim.x + threadIdx.x;   // 0 .. 4095
    if (idx >= 2 * 4 * 32 * 16) return;
    int h    = idx & 15;
    int lane = (idx >> 4) & 31;
    int f    = idx >> 9;            // fragment 0..7
    int t = f >> 2, c = f & 3;
    int kb = (lane < 16) ? 0 : 8;
    int localK = (h < 8) ? (kb + h) : (kb + 16 + (h - 8));
    int K  = 32 * c + localK;
    int kk = K >> 4;
    int i  = K & 15;
    int n  = 16 * t + (lane & 15);
    WfB[idx] = (_Float16)W[(kk * D_IN + i) * D_OUT + n];
}

// ---------------------------------------------------------------------------
// Kernel 2: edge GEMM.  One wave owns a tile of 16 edges.
//   y[e, k*16+i] = basis_k(e) * x[src[e], i]     (formed directly in A regs
//   with packed f16 muls; all 4 K-chunk fragments built up front so the 8
//   WMMAs issue back-to-back with no WAR hazard NOPs)
//   msg = y @ Wf  via 4 K-chunks x 2 N-tiles of v_wmma_f32_16x16x32_f16
//   scatter: global_atomic_add_f32 into agg[dst]
// ---------------------------------------------------------------------------
__global__ void __launch_bounds__(256)
k_edge(const float* __restrict__ x,
       const int*   __restrict__ src,
       const int*   __restrict__ dst,
       const float* __restrict__ attr,
       const _Float16* __restrict__ WfB,
       float* __restrict__ agg)
{
    const int lane   = threadIdx.x & 31;
    const int wave   = (blockIdx.x * blockDim.x + threadIdx.x) >> 5;
    const int nwaves = (gridDim.x * blockDim.x) >> 5;
    const int m  = lane & 15;      // row (edge within tile)
    const int hi = lane >> 4;      // lane half: selects i-range and output rows

    // Persistent B fragments: Wf in exact WMMA B layout (64 VGPRs)
    v16h Bf[2][4];
    {
        const v16h* bp = (const v16h*)WfB;
#pragma unroll
        for (int t = 0; t < 2; ++t)
#pragma unroll
            for (int c = 0; c < 4; ++c)
                Bf[t][c] = bp[(t * 4 + c) * 32 + lane];
    }

    const int ntiles = E_EDGES / 16;   // 200000, exact
    for (int tile = wave; tile < ntiles; tile += nwaves) {
        const int e = tile * 16 + m;

        // streamed-once edge data: non-temporal, keep x/agg hot in L2
        const float p0 = __builtin_nontemporal_load(attr + e * 3 + 0);
        const float p1 = __builtin_nontemporal_load(attr + e * 3 + 1);
        const float p2 = __builtin_nontemporal_load(attr + e * 3 + 2);
        const int   s  = __builtin_nontemporal_load(src + e);

        // degree-1 B-spline basis, K=2 per dim, flat index i0 + 2*i1 + 4*i2
        const float q0 = 1.0f - p0, q1 = 1.0f - p1, q2 = 1.0f - p2;
        float b[8];
        b[0] = q0 * q1 * q2;  b[1] = p0 * q1 * q2;
        b[2] = q0 * p1 * q2;  b[3] = p0 * p1 * q2;
        b[4] = q0 * q1 * p2;  b[5] = p0 * q1 * p2;
        b[6] = q0 * p1 * p2;  b[7] = p0 * p1 * p2;

        // this lane only needs 8 of the 16 input features
        const float4* xp = (const float4*)(x + s * D_IN + hi * 8);
        const float4 x0 = xp[0], x1 = xp[1];
        const float xv[8] = { x0.x, x0.y, x0.z, x0.w, x1.x, x1.y, x1.z, x1.w };

        // one-time f16 conversion: x pairs + broadcast basis pairs
        h2 xh[4];
#pragma unroll
        for (int j = 0; j < 4; ++j) {
            auto p = __builtin_amdgcn_cvt_pkrtz(xv[2 * j], xv[2 * j + 1]);
            __builtin_memcpy(&xh[j], &p, 4);
        }
        h2 bh[8];
#pragma unroll
        for (int k = 0; k < 8; ++k) {
            auto p = __builtin_amdgcn_cvt_pkrtz(b[k], b[k]);
            __builtin_memcpy(&bh[k], &p, 4);
        }

        // prefetch dst rows for the scatter (2 x global_load_b128, NT)
        const int rbase = tile * 16 + hi * 8;
        const v4i* dp = (const v4i*)(dst + rbase);
        const v4i d0 = __builtin_nontemporal_load(dp);
        const v4i d1 = __builtin_nontemporal_load(dp + 1);
        const int dr[8] = { d0.x, d0.y, d0.z, d0.w, d1.x, d1.y, d1.z, d1.w };

        // Build ALL four A fragments first (independent register sets), so
        // the 8 WMMAs below issue back-to-back without WAR-hazard v_nops.
        // halves 0..7  -> kk = 2c   (localK kb..kb+7)
        // halves 8..15 -> kk = 2c+1 (localK kb+16..kb+23)
        union Frag { v16h v; h2 h[8]; };
        Frag A0, A1, A2, A3;
#pragma unroll
        for (int j = 0; j < 4; ++j) {
            A0.h[j]     = bh[0] * xh[j];   // v_pk_mul_f16
            A0.h[4 + j] = bh[1] * xh[j];
            A1.h[j]     = bh[2] * xh[j];
            A1.h[4 + j] = bh[3] * xh[j];
            A2.h[j]     = bh[4] * xh[j];
            A2.h[4 + j] = bh[5] * xh[j];
            A3.h[j]     = bh[6] * xh[j];
            A3.h[4 + j] = bh[7] * xh[j];
        }

        v8f acc0 = {};
        v8f acc1 = {};
        acc0 = __builtin_amdgcn_wmma_f32_16x16x32_f16(
            false, A0.v, false, Bf[0][0], (short)0, acc0, false, false);
        acc1 = __builtin_amdgcn_wmma_f32_16x16x32_f16(
            false, A0.v, false, Bf[1][0], (short)0, acc1, false, false);
        acc0 = __builtin_amdgcn_wmma_f32_16x16x32_f16(
            false, A1.v, false, Bf[0][1], (short)0, acc0, false, false);
        acc1 = __builtin_amdgcn_wmma_f32_16x16x32_f16(
            false, A1.v, false, Bf[1][1], (short)0, acc1, false, false);
        acc0 = __builtin_amdgcn_wmma_f32_16x16x32_f16(
            false, A2.v, false, Bf[0][2], (short)0, acc0, false, false);
        acc1 = __builtin_amdgcn_wmma_f32_16x16x32_f16(
            false, A2.v, false, Bf[1][2], (short)0, acc1, false, false);
        acc0 = __builtin_amdgcn_wmma_f32_16x16x32_f16(
            false, A3.v, false, Bf[0][3], (short)0, acc0, false, false);
        acc1 = __builtin_amdgcn_wmma_f32_16x16x32_f16(
            false, A3.v, false, Bf[1][3], (short)0, acc1, false, false);

        // C layout: VGPR v holds row (v + 8*hi), column = lane&15 (+16 for acc1)
#pragma unroll
        for (int v = 0; v < 8; ++v) {
            atomicAdd(&agg[dr[v] * D_OUT + m],      acc0[v]);
            atomicAdd(&agg[dr[v] * D_OUT + 16 + m], acc1[v]);
        }
    }
}

// ---------------------------------------------------------------------------
// Kernel 2b: edge-count kernel (separate, trivially coalesced)
// ---------------------------------------------------------------------------
__global__ void k_count(const int* __restrict__ dst, float* __restrict__ cnt)
{
    int stride = gridDim.x * blockDim.x;
    for (int i = blockIdx.x * blockDim.x + threadIdx.x; i < E_EDGES; i += stride)
        atomicAdd(&cnt[__builtin_nontemporal_load(dst + i)], 1.0f);
}

// ---------------------------------------------------------------------------
// Kernel 3: per-node: mean-aggregate, root linear, bias, ELU; write pre-BN
// activations to out; block-reduce per-feature sum / sumsq into global accums.
// One thread per (node, feature); 256 threads = 8 nodes per block.
// ---------------------------------------------------------------------------
__global__ void __launch_bounds__(256)
k_node(const float* __restrict__ x,
       const float* __restrict__ Wroot,
       const float* __restrict__ bias,
       const float* __restrict__ agg,
       const float* __restrict__ cnt,
       float* __restrict__ out,
       float* __restrict__ gsum,
       float* __restrict__ gsq)
{
    __shared__ float Wr[D_IN * D_OUT];
    __shared__ float ssum[D_OUT];
    __shared__ float ssq[D_OUT];

    const int tid = threadIdx.x;
    for (int i = tid; i < D_IN * D_OUT; i += blockDim.x) Wr[i] = Wroot[i];
    if (tid < D_OUT) { ssum[tid] = 0.0f; ssq[tid] = 0.0f; }
    __syncthreads();

    const int gid = blockIdx.x * blockDim.x + tid;
    if (gid < N_NODES * D_OUT) {
        const int node = gid >> 5;
        const int n    = gid & 31;
        float c = cnt[node];
        if (c < 1.0f) c = 1.0f;
        float v = agg[gid] / c + bias[n];
        const float* xr = x + node * D_IN;
#pragma unroll
        for (int i = 0; i < D_IN; ++i)
            v = fmaf(xr[i], Wr[i * D_OUT + n], v);
        v = (v > 0.0f) ? v : (expf(v) - 1.0f);   // ELU, alpha = 1
        out[gid] = v;
        atomicAdd(&ssum[n], v);
        atomicAdd(&ssq[n], v * v);
    }
    __syncthreads();
    if (tid < D_OUT) {
        atomicAdd(&gsum[tid], ssum[tid]);
        atomicAdd(&gsq[tid],  ssq[tid]);
    }
}

// ---------------------------------------------------------------------------
// Kernel 4: BatchNorm finalize (in place on out).
// ---------------------------------------------------------------------------
__global__ void __launch_bounds__(256)
k_bn(float* __restrict__ out,
     const float* __restrict__ gsum,
     const float* __restrict__ gsq,
     const float* __restrict__ gamma,
     const float* __restrict__ beta)
{
    const int gid = blockIdx.x * blockDim.x + threadIdx.x;
    if (gid < N_NODES * D_OUT) {
        const int n = gid & 31;
        const float inv_n = 1.0f / (float)N_NODES;
        const float mean  = gsum[n] * inv_n;
        const float var   = gsq[n] * inv_n - mean * mean;
        out[gid] = (out[gid] - mean) * rsqrtf(var + BN_EPS) * gamma[n] + beta[n];
    }
}

// ---------------------------------------------------------------------------
extern "C" void kernel_launch(void* const* d_in, const int* in_sizes, int n_in,
                              void* d_out, int out_size, void* d_ws, size_t ws_size,
                              hipStream_t stream)
{
    const float* x     = (const float*)d_in[0];                 // [N,16]
    const int*   eidx  = (const int*)  d_in[1];                 // [2,E]
    const float* attr  = (const float*)d_in[2];                 // [E,3]
    const float* W     = (const float*)d_in[3];                 // [8,16,32]
    const float* Wroot = (const float*)d_in[4];                 // [16,32]
    const float* bias  = (const float*)d_in[5];                 // [32]
    const float* gamma = (const float*)d_in[6];                 // [32]
    const float* beta  = (const float*)d_in[7];                 // [32]
    float* out = (float*)d_out;

    const int* src = eidx;
    const int* dst = eidx + E_EDGES;

    // workspace layout (floats): agg[N*32] | cnt[N] | gsum[32] | gsq[32] | WfB(f16 x 4096)
    float* ws    = (float*)d_ws;
    float* agg   = ws;
    float* cnt   = agg + (size_t)N_NODES * D_OUT;
    float* gsum  = cnt + N_NODES;
    float* gsq   = gsum + D_OUT;
    _Float16* WfB = (_Float16*)(gsq + D_OUT);   // offset is 32B-aligned for v16h loads

    const int nzero = N_NODES * D_OUT + N_NODES + 2 * D_OUT;

    k_zero<<<2048, 256, 0, stream>>>(ws, nzero);
    k_prep<<<16, 256, 0, stream>>>(W, WfB);
    k_edge<<<1024, 256, 0, stream>>>(x, src, dst, attr, WfB, agg);
    k_count<<<1024, 256, 0, stream>>>(dst, cnt);

    const int nblocks = (N_NODES * D_OUT + 255) / 256;   // 12500
    k_node<<<nblocks, 256, 0, stream>>>(x, Wroot, bias, agg, cnt, out, gsum, gsq);
    k_bn<<<nblocks, 256, 0, stream>>>(out, gsum, gsq, gamma, beta);
}
